// MultiHeadAttention_2637109920104
// MI455X (gfx1250) — matrix operand order repaired and checked by reading, not compiled, not run on hardware
//
#include <hip/hip_runtime.h>

#define BB     4
#define SLEN   2048
#define DMODEL 1024
#define NH     16
#define DHEAD  64
#define LDH    72          // LDS row stride in halfwords (144B = 9*16B, keeps b128 alignment)

typedef __attribute__((ext_vector_type(16))) __bf16 v16bf;
typedef __attribute__((ext_vector_type(8)))  __bf16 v8bf;
typedef __attribute__((ext_vector_type(8)))  float  v8f;

__device__ inline __bf16 f2bf(float f) { return (__bf16)f; }   // v_cvt, RNE

__device__ inline v8f wmma_bf16(v16bf a, v16bf b, v8f c) {
    return __builtin_amdgcn_wmma_f32_16x16x32_bf16(
        false, a, false, b, (short)0, c, false, false);
}

// CDNA5 async global->LDS copy (16B per lane), tracked by ASYNCcnt.
__device__ inline void async_copy_b128(void* lds_dst, const void* gsrc) {
    unsigned off = (unsigned)(size_t)lds_dst;   // flat shared addr: low 32 bits = LDS offset
    asm volatile("global_load_async_to_lds_b128 %0, %1, off"
                 :: "v"(off), "v"(gsrc) : "memory");
}
__device__ inline void wait_async0() {
    asm volatile("s_wait_asynccnt 0" ::: "memory");
}

// A-fragment (16x32 row-major M x K). Two contiguous 8-halfword runs -> 2x b128 loads.
__device__ inline v16bf ld_frag_a(const __bf16* base, int ld, int mrow, int kbase, int lane) {
    const int m = lane & 15, half = lane >> 4;
    const __bf16* p = base + (size_t)(mrow + m) * ld + kbase + half * 8;
    v8bf lo = *(const v8bf*)p;
    v8bf hi = *(const v8bf*)(p + 16);
    return __builtin_shufflevector(lo, hi, 0,1,2,3,4,5,6,7,8,9,10,11,12,13,14,15);
}
// B-fragment (32x16) from an N-major tile: B(k,n) = base[n][k]. Two contiguous 8-halfword runs (same k order as A).
__device__ inline v16bf ld_frag_b(const __bf16* base, int ld, int nrow, int kbase, int lane) {
    const int n = lane & 15, half = lane >> 4;
    const __bf16* p = base + (size_t)(nrow + n) * ld + kbase + half * 8;
    v8bf lo = *(const v8bf*)p;
    v8bf hi = *(const v8bf*)(p + 16);
    return __builtin_shufflevector(lo, hi, 0,1,2,3,4,5,6,7,8,9,10,11,12,13,14,15);
}

// ---------------------------------------------------------------------------
// Prep A: x (f32) -> xb (bf16), elementwise. grid = 4096 x 256, 8 elems/thread.
// ---------------------------------------------------------------------------
__global__ __launch_bounds__(256) void k_cvt(
    const float* __restrict__ in, unsigned short* __restrict__ out)
{
    size_t i = ((size_t)blockIdx.x * 256 + threadIdx.x) * 8;
    float4 a = *(const float4*)(in + i);
    float4 b = *(const float4*)(in + i + 4);
    v8bf p;
    p[0] = f2bf(a.x); p[1] = f2bf(a.y); p[2] = f2bf(a.z); p[3] = f2bf(a.w);
    p[4] = f2bf(b.x); p[5] = f2bf(b.y); p[6] = f2bf(b.z); p[7] = f2bf(b.w);
    *(v8bf*)(out + i) = p;
}

// ---------------------------------------------------------------------------
// Prep B: transpose+convert the 4 weight matrices to N-major bf16 Wt[col][k].
// grid = (16, 16, 4); block = 256; 64x64 tiles through LDS.
// ---------------------------------------------------------------------------
__global__ __launch_bounds__(256) void k_wtrans(
    const float* __restrict__ Wq, const float* __restrict__ Wk,
    const float* __restrict__ Wv, const float* __restrict__ Wo,
    unsigned short* __restrict__ oq, unsigned short* __restrict__ ok,
    unsigned short* __restrict__ ov, unsigned short* __restrict__ oo)
{
    __shared__ __bf16 tile[64][72];
    const int z = blockIdx.z;
    const float* W = (z == 0) ? Wq : (z == 1) ? Wk : (z == 2) ? Wv : Wo;
    unsigned short* O = (z == 0) ? oq : (z == 1) ? ok : (z == 2) ? ov : oo;
    const int rb = blockIdx.y * 64, cb = blockIdx.x * 64;
    const int t = threadIdx.x;
#pragma unroll
    for (int i = 0; i < 16; ++i) {
        int e = t + i * 256, r = e >> 6, c = e & 63;
        tile[r][c] = f2bf(W[(size_t)(rb + r) * DMODEL + cb + c]);
    }
    __syncthreads();
#pragma unroll
    for (int i = 0; i < 16; ++i) {
        int e = t + i * 256, r = e >> 6, c = e & 63;
        O[(size_t)(cb + r) * DMODEL + rb + c] =
            __builtin_bit_cast(unsigned short, tile[c][r]);
    }
}

// ---------------------------------------------------------------------------
// Kernel 1: QKV projection (all-bf16 async-staged WMMA GEMM) + RoPE epilogue.
// grid = (D/64, B*S/128, 3); block = 256. Q/K out: [B,H,S,DK]; V out: [B,H,DK,S].
// ---------------------------------------------------------------------------
__global__ __launch_bounds__(256) void k_qkv_rope(
    const unsigned short* __restrict__ xb, const unsigned short* __restrict__ wqt,
    const unsigned short* __restrict__ wkt, const unsigned short* __restrict__ wvt,
    const int* __restrict__ tp,
    unsigned short* __restrict__ qb, unsigned short* __restrict__ kb,
    unsigned short* __restrict__ vb)
{
    // As[2][128][LDH] (36864B) + Bt[2][64][LDH] (18432B) = 55296B;
    // f32 staging tile Cs[128][66] (33792B) overlays the same region afterwards.
    __shared__ __align__(16) unsigned char smem[55296];
    __bf16 (*As)[128][LDH] = (__bf16 (*)[128][LDH])smem;
    __bf16 (*Bt)[64][LDH]  = (__bf16 (*)[64][LDH])(smem + 36864);
    float  (*Cs)[66]       = (float  (*)[66])smem;

    const int mat = blockIdx.z;
    const unsigned short* Wt = (mat == 0) ? wqt : (mat == 1) ? wkt : wvt;

    const int rowBase = blockIdx.y * 128;
    const int colBase = blockIdx.x * 64;
    const int t = threadIdx.x;
    const int wave = t >> 5, lane = t & 31;
    const int half = lane >> 4, n = lane & 15;
    const int ct = (wave & 3) * 16;        // column sub-tile
    const int wr = (wave >> 2) * 64;       // row quad base (0 or 64)

    v8f acc[4];
#pragma unroll
    for (int ti = 0; ti < 4; ++ti)
#pragma unroll
        for (int i = 0; i < 8; ++i) acc[ti][i] = 0.f;

    auto issue_chunk = [&](int k0, int buf) {
#pragma unroll
        for (int i = 0; i < 4; ++i) {           // As: 128 x 64 halfwords
            int e = t + i * 256, r = e >> 3, c = (e & 7) * 8;
            async_copy_b128(&As[buf][r][c],
                            xb + (size_t)(rowBase + r) * DMODEL + k0 + c);
        }
#pragma unroll
        for (int i = 0; i < 2; ++i) {           // Bt: 64 x 64 halfwords (pre-transposed)
            int e = t + i * 256, r = e >> 3, c = (e & 7) * 8;
            async_copy_b128(&Bt[buf][r][c],
                            Wt + (size_t)(colBase + r) * DMODEL + k0 + c);
        }
    };

    issue_chunk(0, 0);
    wait_async0();
    __syncthreads();
    for (int kc = 0; kc < 16; ++kc) {
        const int cur = kc & 1;
        if (kc < 15) issue_chunk((kc + 1) * 64, cur ^ 1);   // overlap with WMMA
#pragma unroll
        for (int kk = 0; kk < 64; kk += 32) {
            v16bf b = ld_frag_b(&Bt[cur][0][0], LDH, ct, kk, lane);
#pragma unroll
            for (int ti = 0; ti < 4; ++ti) {
                v16bf a = ld_frag_a(&As[cur][0][0], LDH, wr + ti * 16, kk, lane);
                acc[ti] = wmma_bf16(a, b, acc[ti]);
            }
        }
        if (kc < 15) wait_async0();
        __syncthreads();
    }

    // stage f32 results (overlay region; all GEMM reads are behind the barrier)
#pragma unroll
    for (int ti = 0; ti < 4; ++ti)
#pragma unroll
        for (int r = 0; r < 8; ++r)
            Cs[wr + ti * 16 + half * 8 + r][ct + n] = acc[ti][r];
    __syncthreads();

    // RoPE epilogue: the 64-wide tile is exactly one head (DK==64).
    const int h = blockIdx.x;
#pragma unroll
    for (int p = 0; p < 16; ++p) {
        int pi = t * 16 + p;                // 4096 pairs
        int r  = pi >> 5;
        int pr = pi & 31;
        int g  = rowBase + r;
        int bi = g >> 11;                   // /S
        int s  = g & (SLEN - 1);
        int bh = bi * NH + h;
        float v1 = Cs[r][2 * pr], v2 = Cs[r][2 * pr + 1];
        if (mat == 2) {
            // V stored transposed: [B,H,DK,S]
            size_t o = ((size_t)bh * DHEAD + 2 * pr) * SLEN + s;
            vb[o]        = __builtin_bit_cast(unsigned short, f2bf(v1));
            vb[o + SLEN] = __builtin_bit_cast(unsigned short, f2bf(v2));
        } else {
            int pos = tp[s];
            // 10000^(-2*pr/64) = exp2(-pr * log2(10000)/32)
            float inv = exp2f(-0.41524100f * (float)pr);
            float ang = (float)pos * inv;
            float sn, cs;
            __sincosf(ang, &sn, &cs);
            float r1 = v1 * cs - v2 * sn;
            float r2 = v1 * sn + v2 * cs;
            if (mat == 0) { r1 *= 0.125f; r2 *= 0.125f; }   // fold 1/sqrt(DK) into Q
            unsigned short* outp = (mat == 0) ? qb : kb;
            size_t o = ((size_t)bh * SLEN + s) * DHEAD + 2 * pr;
            outp[o]     = __builtin_bit_cast(unsigned short, f2bf(r1));
            outp[o + 1] = __builtin_bit_cast(unsigned short, f2bf(r2));
        }
    }
}

// ---------------------------------------------------------------------------
// Kernel 2: causal flash attention. grid = (S/128, B*H); block = 256.
// Q fragments straight from global; K/V double-buffered via async-to-LDS.
// ---------------------------------------------------------------------------
__global__ __launch_bounds__(256) void k_flash(
    const unsigned short* __restrict__ qbuf, const unsigned short* __restrict__ kbuf,
    const unsigned short* __restrict__ vbuf, unsigned short* __restrict__ Ob)
{
    // Ks[2][64][LDH] + Vt[2][64][LDH] + Ps[128][LDH] = 55296B
    __shared__ __align__(16) unsigned char smem[55296];
    __bf16 (*Ks)[64][LDH] = (__bf16 (*)[64][LDH])smem;
    __bf16 (*Vt)[64][LDH] = (__bf16 (*)[64][LDH])(smem + 18432);
    __bf16 (*Ps)[LDH]     = (__bf16 (*)[LDH])(smem + 36864);

    const int bh = blockIdx.y;
    const int b = bh >> 4, h = bh & 15;
    const int qbase = blockIdx.x * 128;
    const int t = threadIdx.x;
    const int wave = t >> 5, lane = t & 31;
    const int half = lane >> 4, n = lane & 15;
    const int sr = wave * 16;

    const unsigned short* Qp  = qbuf + (size_t)bh * SLEN * DHEAD;   // [s][dk]
    const unsigned short* Kp  = kbuf + (size_t)bh * SLEN * DHEAD;   // [s][dk]
    const unsigned short* VpT = vbuf + (size_t)bh * DHEAD * SLEN;   // [dk][s]

    auto issueKV = [&](int kb0, int buf) {
#pragma unroll
        for (int i = 0; i < 2; ++i) {
            int e = t + i * 256, r = e >> 3, c = (e & 7) * 8;
            async_copy_b128(&Ks[buf][r][c], Kp + (size_t)(kb0 + r) * DHEAD + c);
            async_copy_b128(&Vt[buf][r][c], VpT + (size_t)r * SLEN + kb0 + c);
        }
    };

    const int nj = (qbase >> 6) + 2;
    issueKV(0, 0);
    // per-wave Q fragments directly from global (rows are 128B, 16B-aligned)
    v16bf qa0 = ld_frag_a((const __bf16*)Qp, DHEAD, qbase + sr,  0, lane);
    v16bf qa1 = ld_frag_a((const __bf16*)Qp, DHEAD, qbase + sr, 32, lane);
    wait_async0();
    __syncthreads();

    float m_[8], l_[8];
    v8f oa[4];
#pragma unroll
    for (int r = 0; r < 8; ++r) { m_[r] = -1e30f; l_[r] = 0.f; }
#pragma unroll
    for (int c4 = 0; c4 < 4; ++c4)
#pragma unroll
        for (int r = 0; r < 8; ++r) oa[c4][r] = 0.f;

    for (int j = 0; j < nj; ++j) {
        const int cur = j & 1;
        const int kb0 = j * 64;
        if (j + 1 < nj) issueKV((j + 1) * 64, cur ^ 1);   // overlap with compute

        // wave-uniform causal skip: strip rows are qbase+sr .. qbase+sr+15
        if (kb0 <= qbase + sr + 15) {
            v8f sc[4];
#pragma unroll
            for (int c4 = 0; c4 < 4; ++c4) {
#pragma unroll
                for (int r = 0; r < 8; ++r) sc[c4][r] = 0.f;
                v16bf b0 = ld_frag_b(&Ks[cur][0][0], LDH, c4 * 16,  0, lane);
                sc[c4] = wmma_bf16(qa0, b0, sc[c4]);
                v16bf b1 = ld_frag_b(&Ks[cur][0][0], LDH, c4 * 16, 32, lane);
                sc[c4] = wmma_bf16(qa1, b1, sc[c4]);
            }

            if (kb0 + 63 > qbase + sr) {        // diagonal-crossing tile only
#pragma unroll
                for (int c4 = 0; c4 < 4; ++c4) {
                    int key = kb0 + c4 * 16 + n;
#pragma unroll
                    for (int r = 0; r < 8; ++r) {
                        int q = qbase + sr + half * 8 + r;
                        if (key > q) sc[c4][r] = -1e30f;
                    }
                }
            }

            // online softmax
            float mx[8];
#pragma unroll
            for (int r = 0; r < 8; ++r) {
                float v = fmaxf(fmaxf(sc[0][r], sc[1][r]), fmaxf(sc[2][r], sc[3][r]));
#pragma unroll
                for (int msk = 1; msk < 16; msk <<= 1)
                    v = fmaxf(v, __shfl_xor(v, msk, 32));
                mx[r] = v;
            }
            float sf[8], rs[8];
#pragma unroll
            for (int r = 0; r < 8; ++r) {
                float mn = fmaxf(m_[r], mx[r]);
                sf[r] = __expf(m_[r] - mn);
                m_[r] = mn;
                rs[r] = 0.f;
            }
#pragma unroll
            for (int c4 = 0; c4 < 4; ++c4) {
#pragma unroll
                for (int r = 0; r < 8; ++r) {
                    float pv = __expf(sc[c4][r] - m_[r]);
                    rs[r] += pv;
                    Ps[sr + half * 8 + r][c4 * 16 + n] = f2bf(pv);
                }
            }
#pragma unroll
            for (int r = 0; r < 8; ++r) {
                float v = rs[r];
#pragma unroll
                for (int msk = 1; msk < 16; msk <<= 1)
                    v += __shfl_xor(v, msk, 32);
                l_[r] = l_[r] * sf[r] + v;
            }
#pragma unroll
            for (int c4 = 0; c4 < 4; ++c4)
#pragma unroll
                for (int r = 0; r < 8; ++r) oa[c4][r] *= sf[r];

            // per-wave C-layout -> A-layout handoff: wave-local LDS fence
            asm volatile("s_wait_dscnt 0" ::: "memory");

            v16bf pa0 = ld_frag_a(&Ps[0][0], LDH, sr,  0, lane);
            v16bf pa1 = ld_frag_a(&Ps[0][0], LDH, sr, 32, lane);
#pragma unroll
            for (int c4 = 0; c4 < 4; ++c4) {
                v16bf b0 = ld_frag_b(&Vt[cur][0][0], LDH, c4 * 16,  0, lane);
                oa[c4] = wmma_bf16(pa0, b0, oa[c4]);
                v16bf b1 = ld_frag_b(&Vt[cur][0][0], LDH, c4 * 16, 32, lane);
                oa[c4] = wmma_bf16(pa1, b1, oa[c4]);
            }
        }
        if (j + 1 < nj) wait_async0();
        __syncthreads();
    }

    // normalize and store bf16 attention output [B,S,D]
#pragma unroll
    for (int c4 = 0; c4 < 4; ++c4) {
#pragma unroll
        for (int r = 0; r < 8; ++r) {
            int row = qbase + sr + half * 8 + r;
            int col = h * DHEAD + c4 * 16 + n;
            Ob[((size_t)b * SLEN + row) * DMODEL + col] =
                __builtin_bit_cast(unsigned short, f2bf(oa[c4][r] / l_[r]));
        }
    }
}

// ---------------------------------------------------------------------------
// Kernel 3: output projection O @ Wo (all-bf16 async-staged WMMA GEMM, f32 out).
// grid = (D/64, B*S/128); block = 256.
// ---------------------------------------------------------------------------
__global__ __launch_bounds__(256) void k_proj(
    const unsigned short* __restrict__ Ab, const unsigned short* __restrict__ Wt,
    float* __restrict__ out)
{
    __shared__ __align__(16) unsigned char smem[55296];
    __bf16 (*As)[128][LDH] = (__bf16 (*)[128][LDH])smem;
    __bf16 (*Bt)[64][LDH]  = (__bf16 (*)[64][LDH])(smem + 36864);

    const int rowBase = blockIdx.y * 128;
    const int colBase = blockIdx.x * 64;
    const int t = threadIdx.x;
    const int wave = t >> 5, lane = t & 31;
    const int half = lane >> 4, n = lane & 15;
    const int ct = (wave & 3) * 16;
    const int wr = (wave >> 2) * 64;

    v8f acc[4];
#pragma unroll
    for (int ti = 0; ti < 4; ++ti)
#pragma unroll
        for (int i = 0; i < 8; ++i) acc[ti][i] = 0.f;

    auto issue_chunk = [&](int k0, int buf) {
#pragma unroll
        for (int i = 0; i < 4; ++i) {
            int e = t + i * 256, r = e >> 3, c = (e & 7) * 8;
            async_copy_b128(&As[buf][r][c],
                            Ab + (size_t)(rowBase + r) * DMODEL + k0 + c);
        }
#pragma unroll
        for (int i = 0; i < 2; ++i) {
            int e = t + i * 256, r = e >> 3, c = (e & 7) * 8;
            async_copy_b128(&Bt[buf][r][c],
                            Wt + (size_t)(colBase + r) * DMODEL + k0 + c);
        }
    };

    issue_chunk(0, 0);
    wait_async0();
    __syncthreads();
    for (int kc = 0; kc < 16; ++kc) {
        const int cur = kc & 1;
        if (kc < 15) issue_chunk((kc + 1) * 64, cur ^ 1);
#pragma unroll
        for (int kk = 0; kk < 64; kk += 32) {
            v16bf b = ld_frag_b(&Bt[cur][0][0], LDH, ct, kk, lane);
#pragma unroll
            for (int ti = 0; ti < 4; ++ti) {
                v16bf a = ld_frag_a(&As[cur][0][0], LDH, wr + ti * 16, kk, lane);
                acc[ti] = wmma_bf16(a, b, acc[ti]);
            }
        }
        if (kc < 15) wait_async0();
        __syncthreads();
    }

#pragma unroll
    for (int ti = 0; ti < 4; ++ti)
#pragma unroll
        for (int r = 0; r < 8; ++r)
            out[(size_t)(rowBase + wr + ti * 16 + half * 8 + r) * DMODEL
                + colBase + ct + n] = acc[ti][r];
}

extern "C" void kernel_launch(void* const* d_in, const int* in_sizes, int n_in,
                              void* d_out, int out_size, void* d_ws, size_t ws_size,
                              hipStream_t stream) {
    const float* x  = (const float*)d_in[0];
    const float* Wq = (const float*)d_in[1];
    const float* Wk = (const float*)d_in[2];
    const float* Wv = (const float*)d_in[3];
    const float* Wo = (const float*)d_in[4];
    const int*   tp = (const int*)d_in[5];

    const size_t NX = (size_t)BB * SLEN * DMODEL;       // 8M elements
    const size_t NW = (size_t)DMODEL * DMODEL;          // 1M elements
    unsigned short* xb  = (unsigned short*)d_ws;
    unsigned short* wqt = xb  + NX;
    unsigned short* wkt = wqt + NW;
    unsigned short* wvt = wkt + NW;
    unsigned short* wot = wvt + NW;
    unsigned short* qbu = wot + NW;
    unsigned short* kbu = qbu + NX;
    unsigned short* vbu = kbu + NX;
    unsigned short* obf = vbu + NX;

    dim3 blk(256);
    k_cvt<<<dim3((unsigned)(NX / (256 * 8))), blk, 0, stream>>>(x, xb);
    k_wtrans<<<dim3(16, 16, 4), blk, 0, stream>>>(Wq, Wk, Wv, Wo,
                                                  wqt, wkt, wvt, wot);
    k_qkv_rope<<<dim3(DMODEL / 64, BB * SLEN / 128, 3), blk, 0, stream>>>(
        xb, wqt, wkt, wvt, tp, qbu, kbu, vbu);
    k_flash<<<dim3(SLEN / 128, BB * NH), blk, 0, stream>>>(qbu, kbu, vbu, obf);
    k_proj<<<dim3(DMODEL / 64, BB * SLEN / 128), blk, 0, stream>>>(
        obf, wot, (float*)d_out);
}
